// LRNNBlock_83837761618213
// MI455X (gfx1250) — matrix operand order repaired and checked by reading, not compiled, not run on hardware
//
#include <hip/hip_runtime.h>
#include <hip/hip_bf16.h>
#include <cstdint>

// ---------------------------------------------------------------------------
// h_new = sigmoid(decay) * h + x @ W^T + b        (output duplicated twice)
// x:[16384,2048] f32, W:[2048,2048] f32 (row = out-feature, K contiguous)
//
// bf16x3 split-precision GEMM on v_wmma_f32_16x16x32_bf16 (wave32, gfx1250),
// software-pipelined: next K-tile global loads overlap the WMMA phase.
// ---------------------------------------------------------------------------

#define DIM    2048
#define BATCH  16384
#define BM     128
#define BN     128
#define BK     32
#define LDSS   40      // bf16 elements per LDS row (32 + 8 pad -> 80B stride)

typedef __attribute__((ext_vector_type(16))) __bf16   v16bf;
typedef __attribute__((ext_vector_type(8)))  float    v8f;
typedef __attribute__((ext_vector_type(4)))  float    v4f;
typedef __attribute__((ext_vector_type(8)))  uint16_t v8us;
typedef __attribute__((ext_vector_type(16))) uint16_t v16us;
typedef __attribute__((ext_vector_type(4)))  uint16_t v4us;
typedef __attribute__((ext_vector_type(2)))  uint32_t v2u;

// Packed f32x2 -> bf16x2 (RNE) in one dword; instruction confirmed present in
// gfx1250 codegen (clang emits it for scalar __bf16 casts, half-wasted).
__device__ __forceinline__ uint32_t cvt_pk_bf16(float a, float b) {
    uint32_t r;
    asm("v_cvt_pk_bf16_f32 %0, %1, %2" : "=v"(r) : "v"(a), "v"(b));
    return r;
}

// Split one f32 4-vector into bf16 hi (RNE) and bf16 lo = bf16(f - f32(hi)).
// hi's f32 value recovered from the packed dword: low half u<<16, high half
// u & 0xffff0000 -> ~3 VALU ops per element total.
__device__ __forceinline__ void split_bf16(const v4f v, v4us& hi, v4us& lo) {
    uint32_t u01 = cvt_pk_bf16(v[0], v[1]);
    uint32_t u23 = cvt_pk_bf16(v[2], v[3]);
    float f0 = __builtin_bit_cast(float, u01 << 16);
    float f1 = __builtin_bit_cast(float, u01 & 0xffff0000u);
    float f2 = __builtin_bit_cast(float, u23 << 16);
    float f3 = __builtin_bit_cast(float, u23 & 0xffff0000u);
    uint32_t l01 = cvt_pk_bf16(v[0] - f0, v[1] - f1);
    uint32_t l23 = cvt_pk_bf16(v[2] - f2, v[3] - f3);
    hi = __builtin_bit_cast(v4us, (v2u){u01, u23});
    lo = __builtin_bit_cast(v4us, (v2u){l01, l23});
}

// A fragment 16x32 bf16: lane (0..15) = row, V0..3 hold K=(lane/16)*8 .. +7,
// V4..7 hold K=16+(lane/16)*8 .. +7  -> two ds_load_b128 per fragment.
__device__ __forceinline__ v16bf ld_frag_a(const uint16_t* s, int rowBase, int lane) {
    const int r  = rowBase + (lane & 15);
    const int kh = (lane >> 4) << 3;                    // 0 or 8
    const uint16_t* p = s + r * LDSS + kh;
    v8us lo = *(const v8us*)(p);                        // K = kh..kh+7
    v8us hi = *(const v8us*)(p + 16);                   // K = 16+kh..16+kh+7
    v16us c = __builtin_shufflevector(lo, hi, 0,1,2,3,4,5,6,7,8,9,10,11,12,13,14,15);
    return __builtin_bit_cast(v16bf, c);
}

// B fragment 32x16 bf16: lane (0..15) = column (out-feature), V0..3 hold
// K=(lane/16)*8 .. +7, V4..7 hold K=16+(lane/16)*8 .. +7  -> two ds_load_b128 per fragment.
__device__ __forceinline__ v16bf ld_frag_b(const uint16_t* s, int colBase, int lane) {
    const int e  = colBase + (lane & 15);
    const int kh = (lane >> 4) << 3;                    // 0 or 8
    const uint16_t* p = s + e * LDSS + kh;
    v8us lo = *(const v8us*)(p);                        // K = kh..kh+7
    v8us hi = *(const v8us*)(p + 16);                   // K = 16+kh..16+kh+7
    v16us c = __builtin_shufflevector(lo, hi, 0,1,2,3,4,5,6,7,8,9,10,11,12,13,14,15);
    return __builtin_bit_cast(v16bf, c);
}

__global__ __launch_bounds__(256) void lrnn_wmma_bf16x3_kernel(
    const float* __restrict__ x, const float* __restrict__ h,
    const float* __restrict__ W, const float* __restrict__ b,
    const float* __restrict__ decay, float* __restrict__ out)
{
    __shared__ __align__(16) uint16_t sXh[BM * LDSS];
    __shared__ __align__(16) uint16_t sXl[BM * LDSS];
    __shared__ __align__(16) uint16_t sWh[BN * LDSS];
    __shared__ __align__(16) uint16_t sWl[BN * LDSS];

    const int tid  = threadIdx.x;
    const int lane = tid & 31;
    const int wave = tid >> 5;       // 0..7
    const int wrow = wave >> 2;      // 0..1 : 64-row strip of C
    const int wcol = wave & 3;       // 0..3 : 32-col strip of C

    const int m0 = blockIdx.y * BM;
    const int n0 = blockIdx.x * BN;

    // Per-thread staging coordinates: 4 chunks of float4 covering a 128x32 tile.
    int srow[4], scol[4];
    #pragma unroll
    for (int i = 0; i < 4; ++i) {
        int idx = tid + i * 256;             // 0..1023
        srow[i] = idx >> 3;                  // 0..127
        scol[i] = (idx & 7) << 2;            // 0,4,...,28
    }
    const float* xp = x + (size_t)m0 * DIM;
    const float* Wp = W + (size_t)n0 * DIM;

    v8f acc[4][2];
    #pragma unroll
    for (int i = 0; i < 4; ++i)
        #pragma unroll
        for (int j = 0; j < 2; ++j)
            acc[i][j] = (v8f){0.f,0.f,0.f,0.f,0.f,0.f,0.f,0.f};

    // ---- prologue: preload K-tile 0 into registers ----
    v4f rx[4], rw[4];
    #pragma unroll
    for (int i = 0; i < 4; ++i) {
        rx[i] = *(const v4f*)(xp + (size_t)srow[i] * DIM + scol[i]);
        rw[i] = *(const v4f*)(Wp + (size_t)srow[i] * DIM + scol[i]);
    }

    for (int k0 = 0; k0 < DIM; k0 += BK) {
        __syncthreads();    // previous iteration's fragment ds_loads complete

        // ---- convert registers -> bf16 hi/lo and store to LDS ----
        #pragma unroll
        for (int i = 0; i < 4; ++i) {
            v4us xh, xl, wh, wl;
            split_bf16(rx[i], xh, xl);
            split_bf16(rw[i], wh, wl);
            *(v4us*)(sXh + srow[i] * LDSS + scol[i]) = xh;
            *(v4us*)(sXl + srow[i] * LDSS + scol[i]) = xl;
            *(v4us*)(sWh + srow[i] * LDSS + scol[i]) = wh;
            *(v4us*)(sWl + srow[i] * LDSS + scol[i]) = wl;
        }
        __syncthreads();

        // ---- issue next K-tile global loads; they retire under the WMMAs ----
        if (k0 + BK < DIM) {
            #pragma unroll
            for (int i = 0; i < 4; ++i) {
                rx[i] = *(const v4f*)(xp + (size_t)srow[i] * DIM + (k0 + BK) + scol[i]);
                rw[i] = *(const v4f*)(Wp + (size_t)srow[i] * DIM + (k0 + BK) + scol[i]);
            }
        }

        // ---- load fragments ----
        v16bf Ah[4], Al[4], Bh[2], Bl[2];
        #pragma unroll
        for (int i = 0; i < 4; ++i) {
            Ah[i] = ld_frag_a(sXh, wrow * 64 + i * 16, lane);
            Al[i] = ld_frag_a(sXl, wrow * 64 + i * 16, lane);
        }
        #pragma unroll
        for (int j = 0; j < 2; ++j) {
            Bh[j] = ld_frag_b(sWh, wcol * 32 + j * 16, lane);
            Bl[j] = ld_frag_b(sWl, wcol * 32 + j * 16, lane);
        }

        // ---- bf16x3: hi*hi + hi*lo + lo*hi, f32 accumulate ----
        #pragma unroll
        for (int i = 0; i < 4; ++i) {
            #pragma unroll
            for (int j = 0; j < 2; ++j) {
                acc[i][j] = __builtin_amdgcn_wmma_f32_16x16x32_bf16(
                    false, Ah[i], false, Bh[j], (short)0, acc[i][j], false, false);
                acc[i][j] = __builtin_amdgcn_wmma_f32_16x16x32_bf16(
                    false, Ah[i], false, Bl[j], (short)0, acc[i][j], false, false);
                acc[i][j] = __builtin_amdgcn_wmma_f32_16x16x32_bf16(
                    false, Al[i], false, Bh[j], (short)0, acc[i][j], false, false);
            }
        }
    }

    // ---- epilogue: alpha*h + acc + b, duplicated output ----
    // C/D layout: VGPR v -> M = v (lanes 0-15) or v+8 (lanes 16-31); N = lane%16
    const int ncol = lane & 15;
    const int rsel = (lane >> 4) << 3;
    #pragma unroll
    for (int j = 0; j < 2; ++j) {
        int col = n0 + wcol * 32 + j * 16 + ncol;
        float d     = decay[col];
        float alpha = 1.0f / (1.0f + __expf(-d));
        float bias  = b[col];
        #pragma unroll
        for (int i = 0; i < 4; ++i) {
            int rowBase = m0 + wrow * 64 + i * 16 + rsel;
            #pragma unroll
            for (int v = 0; v < 8; ++v) {
                size_t off = (size_t)(rowBase + v) * DIM + col;
                float r = alpha * h[off] + acc[i][j][v] + bias;
                out[off] = r;
                out[(size_t)BATCH * DIM + off] = r;
            }
        }
    }
}

extern "C" void kernel_launch(void* const* d_in, const int* in_sizes, int n_in,
                              void* d_out, int out_size, void* d_ws, size_t ws_size,
                              hipStream_t stream) {
    (void)in_sizes; (void)n_in; (void)d_ws; (void)ws_size; (void)out_size;
    const float* x     = (const float*)d_in[0];
    const float* h     = (const float*)d_in[1];
    const float* W     = (const float*)d_in[2];
    const float* b     = (const float*)d_in[3];
    const float* decay = (const float*)d_in[4];
    float* out = (float*)d_out;

    dim3 grid(DIM / BN, BATCH / BM);   // (16, 128); x-fastest so 16 blocks share an x tile
    lrnn_wmma_bf16x3_kernel<<<grid, 256, 0, stream>>>(x, h, W, b, decay, out);
}
